// DiffAbXL_23502061044422
// MI455X (gfx1250) — compile-verified, matrix operand order checked by reading, not run on hardware
//
#include <hip/hip_runtime.h>
#include <hip/hip_bf16.h>

typedef __attribute__((ext_vector_type(16))) _Float16 v16h;
typedef __attribute__((ext_vector_type(8)))  _Float16 v8h;
typedef __attribute__((ext_vector_type(8)))  float    v8f;
typedef __attribute__((ext_vector_type(4)))  float    f4v;

#define N_B 2
#define L_S 512
#define RD_ 128
#define PD_ 64
#define H_  12
#define QD_ 32
#define VD_ 32
#define QP_ 8
#define VP_ 8
#define NL_ (N_B * L_S)        // 1024
#define NH_ (N_B * H_)         // 24
#define PROJ_COLS 2016         // 384*3 + 288*3
#define OUT_FEAT 1824          // H*PD + H*VD + H*VP*7
#define INV_SQRT3 0.57735026918962584f
#define QSCALE    0.17677669529663687f  // 1/sqrt(QD)
#define INF_ 100000.0f

// ---------------------------------------------------------------------------
// WMMA fragment helpers (CDNA5 wave32 layouts, CDNA5 ISA §7.12.2)
// A (16x32 f16): lane 0-15 -> M=lane, VGPR v<4 holds K=2v,2v+1, v>=4 holds
// K=16+2(v-4)..; lanes 16-31 add K+8 per half. B (32x16) mirrors with N=lane.
// D (16x16 f32): lane&15 = N, VGPR r -> M = r + 8*(lane>>4).
// ---------------------------------------------------------------------------
__device__ __forceinline__ int frag_k(int lane, int v) {
  int base = (v < 4) ? (v << 1) : (16 + ((v - 4) << 1));
  return base + ((lane & 16) ? 8 : 0);
}

__device__ __forceinline__ v16h load_frag_h(const _Float16* __restrict__ rowp,
                                            int k0, int lane) {
  v16h f;
#pragma unroll
  for (int v = 0; v < 8; ++v) {
    int k = k0 + frag_k(lane, v);
    f[2 * v]     = rowp[k];
    f[2 * v + 1] = rowp[k + 1];
  }
  return f;
}

__device__ __forceinline__ v8f wmma16(v16h a, v16h b, v8f c) {
  return __builtin_amdgcn_wmma_f32_16x16x32_f16(false, a, false, b, (short)0, c,
                                                false, false);
}

__device__ __forceinline__ v8f zero8() {
  v8f z = {0.f, 0.f, 0.f, 0.f, 0.f, 0.f, 0.f, 0.f};
  return z;
}

// ---------------------------------------------------------------------------
// K0a: vectorized f32 -> f16 pack (8 elems/thread: 2x b128 load, 1x b128 store)
// ---------------------------------------------------------------------------
__global__ void k_cvt_h8(const float* __restrict__ s, _Float16* __restrict__ d,
                         int n8) {
  int i = blockIdx.x * blockDim.x + threadIdx.x;
  if (i >= n8) return;
  const f4v* sp = (const f4v*)s + 2 * (size_t)i;
  f4v a = sp[0], b = sp[1];
  v8h o;
#pragma unroll
  for (int k = 0; k < 4; ++k) { o[k] = (_Float16)a[k]; o[4 + k] = (_Float16)b[k]; }
  ((v8h*)d)[i] = o;
}

__global__ void k_zero_h(_Float16* __restrict__ d, int n) {
  int i = blockIdx.x * blockDim.x + threadIdx.x;
  if (i < n) d[i] = (_Float16)0.f;
}

// ---------------------------------------------------------------------------
// K1: fused projection GEMM  raw[1024][2016] = rf_h[1024][128] @ Wall[2016][128]^T
// ---------------------------------------------------------------------------
__global__ __launch_bounds__(32) void k_proj_gemm(
    const _Float16* __restrict__ rf_h, const _Float16* __restrict__ Wall,
    float* __restrict__ raw) {
  int m0 = blockIdx.x * 16, n0 = blockIdx.y * 16;
  int lane = threadIdx.x;
  const _Float16* arow = rf_h + (size_t)(m0 + (lane & 15)) * RD_;
  const _Float16* brow = Wall + (size_t)(n0 + (lane & 15)) * RD_;
  v8f acc = zero8();
#pragma unroll
  for (int kt = 0; kt < 4; ++kt)
    acc = wmma16(load_frag_h(arow, kt * 32, lane),
                 load_frag_h(brow, kt * 32, lane), acc);
  int nn = n0 + (lane & 15), hi = lane >> 4;
#pragma unroll
  for (int r = 0; r < 8; ++r)
    raw[(size_t)(m0 + r + 8 * hi) * PROJ_COLS + nn] = acc[r];
}

// ---------------------------------------------------------------------------
// K2: post-projection elementwise: rotate points to global, build Qext/Kext
// (64-wide, spatial dims pre-scaled by sqrt(2c_h)), VcatT (transposed v||vp),
// and the -c*qq / -c*kk softmax biases.
// ---------------------------------------------------------------------------
__global__ void k_post_proj(const float* __restrict__ raw,
                            const float* __restrict__ R,
                            const float* __restrict__ coord,
                            const float* __restrict__ sc,
                            _Float16* __restrict__ Qext,
                            _Float16* __restrict__ Kext,
                            _Float16* __restrict__ VcatT,
                            float* __restrict__ qbias,
                            float* __restrict__ kbias) {
  int t = blockIdx.x * blockDim.x + threadIdx.x;
  if (t >= NL_ * H_) return;
  int h = t % H_;
  int l = (t / H_) % L_S;
  int n = t / (H_ * L_S);
  int row = n * L_S + l;
  const float* rr = raw + (size_t)row * PROJ_COLS;
  const float* Rm = R + (size_t)row * 9;
  const float* cd = coord + (size_t)row * 3;
  float gamma = log1pf(__expf(sc[h]));
  float c = gamma * (1.0f / 12.0f);  // gamma*sqrt(2/(9*QP))/2, QP=8
  float s = sqrtf(2.0f * c);
  size_t nh = (size_t)(n * H_ + h);
  _Float16* qe = Qext + (nh * L_S + l) * 64;
  _Float16* ke = Kext + (nh * L_S + l) * 64;
#pragma unroll
  for (int d = 0; d < QD_; ++d) {
    qe[d] = (_Float16)(rr[h * QD_ + d] * QSCALE);
    ke[d] = (_Float16)(rr[384 + h * QD_ + d]);
  }
  float qq = 0.f, kk = 0.f;
#pragma unroll
  for (int p = 0; p < QP_; ++p) {
    float ql[3], kl[3];
#pragma unroll
    for (int i = 0; i < 3; ++i) {
      ql[i] = rr[1152 + (h * QP_ + p) * 3 + i];
      kl[i] = rr[1440 + (h * QP_ + p) * 3 + i];
    }
#pragma unroll
    for (int i = 0; i < 3; ++i) {
      float qg = Rm[i * 3] * ql[0] + Rm[i * 3 + 1] * ql[1] + Rm[i * 3 + 2] * ql[2] + cd[i];
      float kg = Rm[i * 3] * kl[0] + Rm[i * 3 + 1] * kl[1] + Rm[i * 3 + 2] * kl[2] + cd[i];
      qq += qg * qg;
      kk += kg * kg;
      qe[32 + p * 3 + i] = (_Float16)(qg * s);
      ke[32 + p * 3 + i] = (_Float16)(kg * s);
    }
  }
#pragma unroll
  for (int d = 56; d < 64; ++d) { qe[d] = (_Float16)0.f; ke[d] = (_Float16)0.f; }
  qbias[nh * L_S + l] = -c * qq;
  kbias[nh * L_S + l] = -c * kk;

  _Float16* vt = VcatT + nh * 64 * L_S;
#pragma unroll
  for (int d = 0; d < VD_; ++d)
    vt[(size_t)d * L_S + l] = (_Float16)rr[768 + h * VD_ + d];
#pragma unroll
  for (int p = 0; p < VP_; ++p) {
    float vl[3];
#pragma unroll
    for (int i = 0; i < 3; ++i) vl[i] = rr[1728 + (h * VP_ + p) * 3 + i];
#pragma unroll
    for (int i = 0; i < 3; ++i) {
      float vg = Rm[i * 3] * vl[0] + Rm[i * 3 + 1] * vl[1] + Rm[i * 3 + 2] * vl[2] + cd[i];
      vt[(size_t)(32 + p * 3 + i) * L_S + l] = (_Float16)vg;
    }
  }
#pragma unroll
  for (int d = 56; d < 64; ++d) vt[(size_t)d * L_S + l] = (_Float16)0.f;
}

// ---------------------------------------------------------------------------
// K3: pair bias GEMM  pairb[nh][i][j] = pair_h[n,i,j,:] @ Wpb[h,:]
// M = flattened pair rows, N = H (padded to 16), K = 64. Reads f16 pair.
// ---------------------------------------------------------------------------
__global__ __launch_bounds__(32) void k_pair_bias(
    const _Float16* __restrict__ pair_h, const _Float16* __restrict__ Wpb_h,
    float* __restrict__ pairb) {
  int p0 = blockIdx.x * 16;
  int lane = threadIdx.x;
  int hcol = lane & 15;
  const _Float16* ap = pair_h + (size_t)(p0 + (lane & 15)) * PD_;
  __builtin_prefetch(ap + 16 * PD_, 0, 0);
  const _Float16* bp = Wpb_h + hcol * PD_;  // rows 12..15 are zero-padded
  v8f acc = zero8();
#pragma unroll
  for (int kt = 0; kt < 2; ++kt)
    acc = wmma16(load_frag_h(ap, kt * 32, lane),
                 load_frag_h(bp, kt * 32, lane), acc);
  if (hcol < H_) {
#pragma unroll
    for (int r = 0; r < 8; ++r) {
      int m = p0 + r + 8 * (lane >> 4);
      int n = m >> 18;
      int rem = m & ((1 << 18) - 1);
      int i = rem >> 9, j = rem & (L_S - 1);
      pairb[((size_t)(n * H_ + hcol) * L_S + i) * L_S + j] = acc[r];
    }
  }
}

// ---------------------------------------------------------------------------
// K4: fused logits (K=64 WMMA) + bias + mask + row softmax -> alpha (f16)
// One block = (n,h, 16 query rows). 4 waves cover the 32 j-tiles.
// ---------------------------------------------------------------------------
__global__ __launch_bounds__(128) void k_attn_softmax(
    const _Float16* __restrict__ Qext, const _Float16* __restrict__ Kext,
    const float* __restrict__ qbias, const float* __restrict__ kbias,
    const float* __restrict__ pairb, const unsigned char* __restrict__ mask,
    _Float16* __restrict__ alpha) {
  __shared__ float slog[16 * L_S];
  __shared__ float sred[16 * 8];
  __shared__ float srow[16];
  int b = blockIdx.x;
  int itile = b & 31, nh = b >> 5;
  int n = nh / H_;
  int i0 = itile * 16;
  int tid = threadIdx.x, wave = tid >> 5, lane = tid & 31;
  const _Float16* qbase = Qext + (size_t)nh * L_S * 64;
  const _Float16* kbase = Kext + (size_t)nh * L_S * 64;
  const float* qb = qbias + (size_t)nh * L_S;
  const float* kb = kbias + (size_t)nh * L_S;
  const float* pb = pairb + (size_t)nh * L_S * L_S;
  const _Float16* arow = qbase + (size_t)(i0 + (lane & 15)) * 64;
  v16h a0 = load_frag_h(arow, 0, lane);
  v16h a1 = load_frag_h(arow, 32, lane);
  for (int jt = wave * 8; jt < wave * 8 + 8; ++jt) {
    int j0 = jt * 16;
    const _Float16* brow = kbase + (size_t)(j0 + (lane & 15)) * 64;
    v8f acc = zero8();
    acc = wmma16(a0, load_frag_h(brow, 0, lane), acc);
    acc = wmma16(a1, load_frag_h(brow, 32, lane), acc);
    int j = j0 + (lane & 15);
    float kbj = kb[j];
    int mj = mask[n * L_S + j];
#pragma unroll
    for (int r = 0; r < 8; ++r) {
      int m = r + 8 * (lane >> 4);
      int i = i0 + m;
      float v = acc[r] + qb[i] + kbj + pb[(size_t)i * L_S + j];
      v *= INV_SQRT3;
      if (!(mask[n * L_S + i] && mj)) v -= INF_;
      slog[m * L_S + j] = v;
    }
  }
  __syncthreads();
  int row = tid >> 3, sub = tid & 7;
  float mx = -3.0e38f;
  for (int j = sub * 64; j < sub * 64 + 64; ++j) mx = fmaxf(mx, slog[row * L_S + j]);
  sred[row * 8 + sub] = mx;
  __syncthreads();
  if (sub == 0) {
    float m2 = sred[row * 8];
    for (int q = 1; q < 8; ++q) m2 = fmaxf(m2, sred[row * 8 + q]);
    srow[row] = m2;
  }
  __syncthreads();
  float rm = srow[row];
  float sm = 0.f;
  for (int j = sub * 64; j < sub * 64 + 64; ++j) {
    float e = __expf(slog[row * L_S + j] - rm);
    slog[row * L_S + j] = e;
    sm += e;
  }
  sred[row * 8 + sub] = sm;
  __syncthreads();
  if (sub == 0) {
    float s2 = 0.f;
    for (int q = 0; q < 8; ++q) s2 += sred[row * 8 + q];
    srow[row] = 1.0f / s2;
  }
  __syncthreads();
  int i = i0 + row;
  float scale = srow[row] * (mask[n * L_S + i] ? 1.0f : 0.0f);
  _Float16* ar = alpha + ((size_t)nh * L_S + i) * L_S;
  for (int j = sub * 64; j < sub * 64 + 64; ++j)
    ar[j] = (_Float16)(slog[row * L_S + j] * scale);
}

// ---------------------------------------------------------------------------
// K5: na_out[nh][i][0:64] = alpha[nh] (LxL) @ [V || vp_global] (Lx64), K=512
// ---------------------------------------------------------------------------
__global__ __launch_bounds__(32) void k_attn_apply(
    const _Float16* __restrict__ alpha, const _Float16* __restrict__ VcatT,
    float* __restrict__ na_out) {
  int nh = blockIdx.x;
  int i0 = blockIdx.y * 16, c0 = blockIdx.z * 16;
  int lane = threadIdx.x;
  const _Float16* arow = alpha + ((size_t)nh * L_S + (i0 + (lane & 15))) * L_S;
  const _Float16* brow = VcatT + ((size_t)nh * 64 + (c0 + (lane & 15))) * L_S;
  v8f acc = zero8();
  for (int kt = 0; kt < 16; ++kt)
    acc = wmma16(load_frag_h(arow, kt * 32, lane),
                 load_frag_h(brow, kt * 32, lane), acc);
  int cc = c0 + (lane & 15);
#pragma unroll
  for (int r = 0; r < 8; ++r)
    na_out[((size_t)nh * L_S + i0 + r + 8 * (lane >> 4)) * 64 + cc] = acc[r];
}

// ---------------------------------------------------------------------------
// K6: feat_pair per (n,i): alpha^T (Hx512, padded to 16) @ pair_h[n,i] (512x64)
// Padded lanes read a shared zero row (no per-element selects). f16 pair hits
// L2 (67 MB < 192 MB) on this second pass.
// ---------------------------------------------------------------------------
__global__ __launch_bounds__(32) void k_feat_pair(
    const _Float16* __restrict__ alpha, const _Float16* __restrict__ pair_h,
    const _Float16* __restrict__ zrow, float* __restrict__ featp) {
  int row = blockIdx.x;               // n*L + i
  int n = row >> 9, i = row & (L_S - 1);
  int c0 = blockIdx.y * 16;
  int lane = threadIdx.x;
  int h = lane & 15;
  const _Float16* arow =
      (h < H_) ? alpha + ((size_t)(n * H_ + h) * L_S + i) * L_S : zrow;
  const _Float16* pbase = pair_h + (size_t)row * L_S * PD_;
  int cc = c0 + (lane & 15);
  v8f acc = zero8();
  for (int kt = 0; kt < 16; ++kt) {
    int k0 = kt * 32;
    __builtin_prefetch(pbase + (size_t)(k0 + 32 + frag_k(lane, 0)) * PD_ + cc, 0, 0);
    v16h a = load_frag_h(arow, k0, lane);
    v16h b;
#pragma unroll
    for (int v = 0; v < 8; ++v) {
      int k = k0 + frag_k(lane, v);
      b[2 * v]     = pbase[(size_t)k * PD_ + cc];
      b[2 * v + 1] = pbase[(size_t)(k + 1) * PD_ + cc];
    }
    acc = wmma16(a, b, acc);
  }
#pragma unroll
  for (int r = 0; r < 8; ++r) {
    int hh = r + 8 * (lane >> 4);
    if (hh < H_) featp[((size_t)row * H_ + hh) * PD_ + cc] = acc[r];
  }
}

// ---------------------------------------------------------------------------
// K7: assemble feat_all[1024][1824] (f16): pair | node | points | dist | dir
// ---------------------------------------------------------------------------
__global__ void k_assemble(const float* __restrict__ na_out,
                           const float* __restrict__ featp,
                           const float* __restrict__ R,
                           const float* __restrict__ coord,
                           _Float16* __restrict__ feat_all) {
  int t = blockIdx.x * blockDim.x + threadIdx.x;
  if (t >= NL_ * H_) return;
  int h = t % H_;
  int l = (t / H_) % L_S;
  int n = t / (H_ * L_S);
  int row = n * L_S + l;
  const float* Rm = R + (size_t)row * 9;
  const float* cd = coord + (size_t)row * 3;
  const float* na = na_out + ((size_t)(n * H_ + h) * L_S + l) * 64;
  const float* fp = featp + ((size_t)row * H_ + h) * PD_;
  _Float16* fa = feat_all + (size_t)row * OUT_FEAT;
#pragma unroll
  for (int c = 0; c < PD_; ++c) fa[h * PD_ + c] = (_Float16)fp[c];
#pragma unroll
  for (int d = 0; d < VD_; ++d) fa[768 + h * VD_ + d] = (_Float16)na[d];
#pragma unroll
  for (int p = 0; p < VP_; ++p) {
    float df[3], ft[3];
#pragma unroll
    for (int i = 0; i < 3; ++i) df[i] = na[32 + p * 3 + i] - cd[i];
#pragma unroll
    for (int i = 0; i < 3; ++i)  // R^T * diff  ('nlji')
      ft[i] = Rm[0 * 3 + i] * df[0] + Rm[1 * 3 + i] * df[1] + Rm[2 * 3 + i] * df[2];
    float dist = sqrtf(ft[0] * ft[0] + ft[1] * ft[1] + ft[2] * ft[2]);
    float invd = 1.0f / (dist + 1e-4f);
#pragma unroll
    for (int i = 0; i < 3; ++i) {
      fa[1152 + h * 24 + p * 3 + i] = (_Float16)ft[i];
      fa[1536 + h * 24 + p * 3 + i] = (_Float16)(ft[i] * invd);
    }
    fa[1440 + h * VP_ + p] = (_Float16)dist;
  }
}

// ---------------------------------------------------------------------------
// K8: out = feat_all @ Wout^T + bout; mask; + residue_feat; LayerNorm1.
// 8 waves per block each own one 16-col tile of a 16-row block; LN via LDS.
// ---------------------------------------------------------------------------
__global__ __launch_bounds__(256) void k_outproj_ln(
    const _Float16* __restrict__ feat_all, const _Float16* __restrict__ Wout_h,
    const float* __restrict__ bout, const float* __restrict__ rf,
    const unsigned char* __restrict__ mask, const float* __restrict__ g,
    const float* __restrict__ bb, float* __restrict__ x,
    _Float16* __restrict__ x_h) {
  __shared__ float xs[16 * RD_];
  int m0 = blockIdx.x * 16;
  int tid = threadIdx.x, wave = tid >> 5, lane = tid & 31;
  int c0 = wave * 16;
  const _Float16* arow = feat_all + (size_t)(m0 + (lane & 15)) * OUT_FEAT;
  const _Float16* brow = Wout_h + (size_t)(c0 + (lane & 15)) * OUT_FEAT;
  v8f acc = zero8();
  for (int kt = 0; kt < 57; ++kt)
    acc = wmma16(load_frag_h(arow, kt * 32, lane),
                 load_frag_h(brow, kt * 32, lane), acc);
  int cc = c0 + (lane & 15);
#pragma unroll
  for (int r = 0; r < 8; ++r) {
    int m = r + 8 * (lane >> 4);
    int rowg = m0 + m;
    float v = acc[r] + bout[cc];
    if (!mask[rowg]) v = 0.0f;
    xs[m * RD_ + cc] = rf[(size_t)rowg * RD_ + cc] + v;
  }
  __syncthreads();
  if (tid < 16) {
    int rowg = m0 + tid;
    float mn = 0.f;
    for (int c = 0; c < RD_; ++c) mn += xs[tid * RD_ + c];
    mn *= (1.0f / RD_);
    float var = 0.f;
    for (int c = 0; c < RD_; ++c) {
      float d = xs[tid * RD_ + c] - mn;
      var += d * d;
    }
    float inv = rsqrtf(var * (1.0f / RD_) + 1e-5f);
    for (int c = 0; c < RD_; ++c) {
      float v = (xs[tid * RD_ + c] - mn) * inv * g[c] + bb[c];
      x[(size_t)rowg * RD_ + c] = v;
      x_h[(size_t)rowg * RD_ + c] = (_Float16)v;
    }
  }
}

// ---------------------------------------------------------------------------
// K9: MLP GEMM 1024x128x128 (+bias, optional relu) -> f16
// ---------------------------------------------------------------------------
__global__ __launch_bounds__(256) void k_mlp(const _Float16* __restrict__ in_h,
                                             const _Float16* __restrict__ W_h,
                                             const float* __restrict__ bias,
                                             _Float16* __restrict__ out_h,
                                             int relu) {
  int m0 = blockIdx.x * 16;
  int tid = threadIdx.x, wave = tid >> 5, lane = tid & 31;
  int c0 = wave * 16;
  const _Float16* arow = in_h + (size_t)(m0 + (lane & 15)) * RD_;
  const _Float16* brow = W_h + (size_t)(c0 + (lane & 15)) * RD_;
  v8f acc = zero8();
#pragma unroll
  for (int kt = 0; kt < 4; ++kt)
    acc = wmma16(load_frag_h(arow, kt * 32, lane),
                 load_frag_h(brow, kt * 32, lane), acc);
  int cc = c0 + (lane & 15);
#pragma unroll
  for (int r = 0; r < 8; ++r) {
    int rowg = m0 + r + 8 * (lane >> 4);
    float v = acc[r] + bias[cc];
    if (relu) v = fmaxf(v, 0.0f);
    out_h[(size_t)rowg * RD_ + cc] = (_Float16)v;
  }
}

// ---------------------------------------------------------------------------
// K10: last MLP GEMM + residual + LayerNorm2 -> d_out (f32)
// ---------------------------------------------------------------------------
__global__ __launch_bounds__(256) void k_mlp3_ln(
    const _Float16* __restrict__ h2, const _Float16* __restrict__ Wm3_h,
    const float* __restrict__ bm3, const float* __restrict__ x,
    const float* __restrict__ g, const float* __restrict__ bb,
    float* __restrict__ out) {
  __shared__ float xs[16 * RD_];
  int m0 = blockIdx.x * 16;
  int tid = threadIdx.x, wave = tid >> 5, lane = tid & 31;
  int c0 = wave * 16;
  const _Float16* arow = h2 + (size_t)(m0 + (lane & 15)) * RD_;
  const _Float16* brow = Wm3_h + (size_t)(c0 + (lane & 15)) * RD_;
  v8f acc = zero8();
#pragma unroll
  for (int kt = 0; kt < 4; ++kt)
    acc = wmma16(load_frag_h(arow, kt * 32, lane),
                 load_frag_h(brow, kt * 32, lane), acc);
  int cc = c0 + (lane & 15);
#pragma unroll
  for (int r = 0; r < 8; ++r) {
    int m = r + 8 * (lane >> 4);
    int rowg = m0 + m;
    xs[m * RD_ + cc] = x[(size_t)rowg * RD_ + cc] + acc[r] + bm3[cc];
  }
  __syncthreads();
  if (tid < 16) {
    int rowg = m0 + tid;
    float mn = 0.f;
    for (int c = 0; c < RD_; ++c) mn += xs[tid * RD_ + c];
    mn *= (1.0f / RD_);
    float var = 0.f;
    for (int c = 0; c < RD_; ++c) {
      float d = xs[tid * RD_ + c] - mn;
      var += d * d;
    }
    float inv = rsqrtf(var * (1.0f / RD_) + 1e-5f);
    for (int c = 0; c < RD_; ++c)
      out[(size_t)rowg * RD_ + c] = (xs[tid * RD_ + c] - mn) * inv * g[c] + bb[c];
  }
}

// ---------------------------------------------------------------------------
// Host launcher
// ---------------------------------------------------------------------------
static inline size_t ws_take(size_t& off, size_t bytes) {
  size_t cur = (off + 255) & ~(size_t)255;
  off = cur + bytes;
  return cur;
}

extern "C" void kernel_launch(void* const* d_in, const int* in_sizes, int n_in,
                              void* d_out, int out_size, void* d_ws, size_t ws_size,
                              hipStream_t stream) {
  (void)in_sizes; (void)n_in; (void)out_size; (void)ws_size;
  const float* R      = (const float*)d_in[0];
  const float* coord  = (const float*)d_in[1];
  const float* rf     = (const float*)d_in[2];
  const float* pair   = (const float*)d_in[3];
  const unsigned char* mask = (const unsigned char*)d_in[4];
  const float* Wq   = (const float*)d_in[5];
  const float* Wk   = (const float*)d_in[6];
  const float* Wv   = (const float*)d_in[7];
  const float* Wpb  = (const float*)d_in[8];
  const float* sc   = (const float*)d_in[9];
  const float* Wqp  = (const float*)d_in[10];
  const float* Wkp  = (const float*)d_in[11];
  const float* Wvp  = (const float*)d_in[12];
  const float* Wout = (const float*)d_in[13];
  const float* bout = (const float*)d_in[14];
  const float* Wm1  = (const float*)d_in[15];
  const float* bm1  = (const float*)d_in[16];
  const float* Wm2  = (const float*)d_in[17];
  const float* bm2  = (const float*)d_in[18];
  const float* Wm3  = (const float*)d_in[19];
  const float* bm3  = (const float*)d_in[20];
  const float* ln1g = (const float*)d_in[21];
  const float* ln1b = (const float*)d_in[22];
  const float* ln2g = (const float*)d_in[23];
  const float* ln2b = (const float*)d_in[24];
  float* outp = (float*)d_out;
  char* ws = (char*)d_ws;

  size_t off = 0;
  _Float16* rf_h   = (_Float16*)(ws + ws_take(off, (size_t)NL_ * RD_ * 2));
  _Float16* Wall_h = (_Float16*)(ws + ws_take(off, (size_t)PROJ_COLS * RD_ * 2));
  _Float16* Wpb_h  = (_Float16*)(ws + ws_take(off, (size_t)16 * PD_ * 2));
  _Float16* Wout_h = (_Float16*)(ws + ws_take(off, (size_t)RD_ * OUT_FEAT * 2));
  _Float16* Wm1_h  = (_Float16*)(ws + ws_take(off, (size_t)RD_ * RD_ * 2));
  _Float16* Wm2_h  = (_Float16*)(ws + ws_take(off, (size_t)RD_ * RD_ * 2));
  _Float16* Wm3_h  = (_Float16*)(ws + ws_take(off, (size_t)RD_ * RD_ * 2));
  _Float16* pair_h = (_Float16*)(ws + ws_take(off, (size_t)N_B * L_S * L_S * PD_ * 2));
  _Float16* zrow   = (_Float16*)(ws + ws_take(off, (size_t)L_S * 2));
  float*    raw    = (float*)(ws + ws_take(off, (size_t)NL_ * PROJ_COLS * 4));
  _Float16* Qext   = (_Float16*)(ws + ws_take(off, (size_t)NH_ * L_S * 64 * 2));
  _Float16* Kext   = (_Float16*)(ws + ws_take(off, (size_t)NH_ * L_S * 64 * 2));
  _Float16* VcatT  = (_Float16*)(ws + ws_take(off, (size_t)NH_ * 64 * L_S * 2));
  float*    qbias  = (float*)(ws + ws_take(off, (size_t)NH_ * L_S * 4));
  float*    kbias  = (float*)(ws + ws_take(off, (size_t)NH_ * L_S * 4));
  float*    pairb  = (float*)(ws + ws_take(off, (size_t)NH_ * L_S * L_S * 4));
  _Float16* alpha  = (_Float16*)(ws + ws_take(off, (size_t)NH_ * L_S * L_S * 2));
  float*    na_out = (float*)(ws + ws_take(off, (size_t)NH_ * L_S * 64 * 4));
  float*    featp  = (float*)(ws + ws_take(off, (size_t)NL_ * H_ * PD_ * 4));
  _Float16* feat_a = (_Float16*)(ws + ws_take(off, (size_t)NL_ * OUT_FEAT * 2));
  float*    xbuf   = (float*)(ws + ws_take(off, (size_t)NL_ * RD_ * 4));
  _Float16* x_h    = (_Float16*)(ws + ws_take(off, (size_t)NL_ * RD_ * 2));
  _Float16* h1_h   = (_Float16*)(ws + ws_take(off, (size_t)NL_ * RD_ * 2));
  _Float16* h2_h   = (_Float16*)(ws + ws_take(off, (size_t)NL_ * RD_ * 2));

  auto cvt8 = [&](const float* s, _Float16* d, size_t n) {
    int n8 = (int)(n / 8);
    k_cvt_h8<<<(n8 + 255) / 256, 256, 0, stream>>>(s, d, n8);
  };
  // Pack weights (concatenated Wall rows: q | k | v | qp | kp | vp)
  cvt8(Wq,  Wall_h + 0 * 128,     (size_t)384 * 128);
  cvt8(Wk,  Wall_h + 384 * 128,   (size_t)384 * 128);
  cvt8(Wv,  Wall_h + 768 * 128,   (size_t)384 * 128);
  cvt8(Wqp, Wall_h + 1152 * 128,  (size_t)288 * 128);
  cvt8(Wkp, Wall_h + 1440 * 128,  (size_t)288 * 128);
  cvt8(Wvp, Wall_h + 1728 * 128,  (size_t)288 * 128);
  cvt8(rf,  rf_h,  (size_t)NL_ * RD_);
  k_zero_h<<<(16 * PD_ + 255) / 256, 256, 0, stream>>>(Wpb_h, 16 * PD_);
  cvt8(Wpb, Wpb_h, (size_t)H_ * PD_);
  cvt8(Wout, Wout_h, (size_t)RD_ * OUT_FEAT);
  cvt8(Wm1, Wm1_h, (size_t)RD_ * RD_);
  cvt8(Wm2, Wm2_h, (size_t)RD_ * RD_);
  cvt8(Wm3, Wm3_h, (size_t)RD_ * RD_);
  cvt8(pair, pair_h, (size_t)N_B * L_S * L_S * PD_);
  k_zero_h<<<(L_S + 255) / 256, 256, 0, stream>>>(zrow, L_S);

  // 1. fused projection GEMM
  k_proj_gemm<<<dim3(NL_ / 16, PROJ_COLS / 16), 32, 0, stream>>>(rf_h, Wall_h, raw);
  // 2. rotate / build attention operands
  k_post_proj<<<(NL_ * H_ + 255) / 256, 256, 0, stream>>>(
      raw, R, coord, sc, Qext, Kext, VcatT, qbias, kbias);
  // 3. pair bias
  k_pair_bias<<<(N_B * L_S * L_S) / 16, 32, 0, stream>>>(pair_h, Wpb_h, pairb);
  // 4. logits + softmax
  k_attn_softmax<<<NH_ * (L_S / 16), 128, 0, stream>>>(
      Qext, Kext, qbias, kbias, pairb, mask, alpha);
  // 5. alpha @ [V || vp]
  k_attn_apply<<<dim3(NH_, L_S / 16, 4), 32, 0, stream>>>(alpha, VcatT, na_out);
  // 6. alpha^T @ pair
  k_feat_pair<<<dim3(NL_, PD_ / 16), 32, 0, stream>>>(alpha, pair_h, zrow, featp);
  // 7. assemble 1824-wide features
  k_assemble<<<(NL_ * H_ + 255) / 256, 256, 0, stream>>>(na_out, featp, R, coord, feat_a);
  // 8. output projection + mask + residual + LN1
  k_outproj_ln<<<NL_ / 16, 256, 0, stream>>>(feat_a, Wout_h, bout, rf, mask,
                                             ln1g, ln1b, xbuf, x_h);
  // 9. MLP
  k_mlp<<<NL_ / 16, 256, 0, stream>>>(x_h, Wm1_h, bm1, h1_h, 1);
  k_mlp<<<NL_ / 16, 256, 0, stream>>>(h1_h, Wm2_h, bm2, h2_h, 1);
  k_mlp3_ln<<<NL_ / 16, 256, 0, stream>>>(h2_h, Wm3_h, bm3, xbuf, ln2g, ln2b, outp);
}
